// NBF_48095043780813
// MI455X (gfx1250) — matrix operand order repaired and checked by reading, not compiled, not run on hardware
//
#include <hip/hip_runtime.h>

// NBFNet-cluttr forward for MI455X (gfx1250), wave32.
// Workload is L2-resident gather/scatter bound (~116 MB working set vs 192 MB L2);
// the cosine-sim GEMM [E x 128] x [128 x 32] is done with v_wmma_f32_16x16x32_f16.

#define N_NODES 50000
#define H       128
#define P       18
#define PPAD    32
#define EPSN    1e-12f
#define EPSE    1e-8f

typedef __attribute__((ext_vector_type(16))) _Float16 v16h;
typedef __attribute__((ext_vector_type(8)))  _Float16 v8h;
typedef __attribute__((ext_vector_type(8)))  float    v8f;

// Monotone float -> uint encoding so scatter-max needs only integer atomicMax.
__device__ __forceinline__ unsigned fenc(float x) {
    unsigned b = __float_as_uint(x);
    return (b & 0x80000000u) ? ~b : (b | 0x80000000u);
}
__device__ __forceinline__ float fdec(unsigned u) {
    return (u & 0x80000000u) ? __uint_as_float(u & 0x7fffffffu)
                             : __uint_as_float(~u);
}

__global__ void init_hidden_kernel(float* __restrict__ hidden,
                                   const float* __restrict__ srcEmb, int NH) {
    int i = blockIdx.x * blockDim.x + threadIdx.x;
    if (i < NH) hidden[i] = (i < H) ? srcEmb[i] : 0.0f;
}

__global__ void layer_reset_kernel(float* __restrict__ hout,
                                   unsigned* __restrict__ segmaxE,
                                   float* __restrict__ segsum, int NH, int N) {
    int i = blockIdx.x * blockDim.x + threadIdx.x;
    if (i < NH) hout[i] = 0.0f;
    if (i < N) { segmaxE[i] = 0u; segsum[i] = 0.0f; }  // 0u decodes below any real logit
}

// Pass 1: per edge, gather+normalize hidden[src], WMMA cosine sims vs normalized
// prototypes, softmax over P, entropy -> logit, atomic segment-max over dst.
// 256 threads = 8 waves; each wave owns 16 edges (one 16x32 WMMA output tile pair).
__global__ __launch_bounds__(256, 2)
void edge_sims_kernel(const float* __restrict__ hidden_in,
                      const float* __restrict__ proto,      // [P][H], this layer
                      const int* __restrict__ src,
                      const int* __restrict__ dst,
                      float* __restrict__ proba,            // [E][P]
                      float* __restrict__ logits,           // [E]
                      unsigned* __restrict__ segmaxE,       // [N] encoded
                      int E) {
    __shared__ __align__(64) _Float16 protoH[PPAD][H];      // normalized protos, f16
    __shared__ __align__(64) _Float16 aStage[8][16][H];     // per-wave A rows, f16
    __shared__ float simsS[8][16][PPAD];

    const int tid = threadIdx.x;

    // Normalize prototypes once per block (one thread per padded row).
    if (tid < PPAD) {
        if (tid < P) {
            float ss = 0.0f;
            #pragma unroll 4
            for (int k = 0; k < H; ++k) { float v = proto[tid * H + k]; ss += v * v; }
            const float sc = 1.0f / fmaxf(sqrtf(ss), EPSN);
            for (int k = 0; k < H; ++k)
                protoH[tid][k] = (_Float16)(proto[tid * H + k] * sc);
        } else {
            for (int k = 0; k < H; ++k) protoH[tid][k] = (_Float16)0.0f;
        }
    }
    __syncthreads();

    const int w        = tid >> 5;
    const int lane     = tid & 31;
    const int hi       = lane >> 4;      // half-wave
    const int nlo      = lane & 15;
    const int edgeBase = blockIdx.x * 128 + w * 16;

    // Gather 16 hidden rows (512B each, L2-resident) and L2-normalize into LDS f16.
    for (int e16 = 0; e16 < 16; ++e16) {
        const int e = edgeBase + e16;
        float x0 = 0, x1 = 0, x2 = 0, x3 = 0;
        if (e < E) {
            const float* hp = hidden_in + (size_t)src[e] * H + lane * 4;
            x0 = hp[0]; x1 = hp[1]; x2 = hp[2]; x3 = hp[3];
        }
        float ss = x0 * x0 + x1 * x1 + x2 * x2 + x3 * x3;
        #pragma unroll
        for (int off = 16; off > 0; off >>= 1) ss += __shfl_xor(ss, off, 32);
        const float sc = 1.0f / fmaxf(sqrtf(ss), EPSN);
        _Float16* ap = &aStage[w][e16][lane * 4];
        ap[0] = (_Float16)(x0 * sc); ap[1] = (_Float16)(x1 * sc);
        ap[2] = (_Float16)(x2 * sc); ap[3] = (_Float16)(x3 * sc);
    }
    // Same-wave LDS ops are in order; no barrier needed (aStage is wave-private).

    // WMMA: D[16x32] = A[16x128] x protoH^T, K split in 4 blocks of 32.
    v8f c0 = {}; v8f c1 = {};
    #pragma unroll
    for (int kb = 0; kb < 4; ++kb) {
        // 16-bit A fragment layout: lane m=nlo; K = kb*32 + 16*hi .. +15 (contiguous)
        const v16h* arow = (const v16h*)&aStage[w][nlo][kb * 32];
        v16h a  = arow[hi];
        // 16-bit B fragment: lane = column nlo; K = kb*32 + 16*hi .. +15 (contiguous)
        const v16h* b0p = (const v16h*)&protoH[nlo][kb * 32];
        const v16h* b1p = (const v16h*)&protoH[16 + nlo][kb * 32];
        v16h b0 = b0p[hi];
        v16h b1 = b1p[hi];
        c0 = __builtin_amdgcn_wmma_f32_16x16x32_f16(false, a, false, b0,
                                                    (short)0, c0, false, false);
        c1 = __builtin_amdgcn_wmma_f32_16x16x32_f16(false, a, false, b1,
                                                    (short)0, c1, false, false);
    }

    // C/D layout: VGPR v, lanes 0-15 -> M=v, lanes 16-31 -> M=v+8; N = nlo.
    #pragma unroll
    for (int v = 0; v < 8; ++v) {
        simsS[w][v + 8 * hi][nlo]      = c0[v];
        simsS[w][v + 8 * hi][16 + nlo] = c1[v];
    }

    // Softmax over P, entropy, logit; one lane per edge.
    if (lane < 16) {
        const int e = edgeBase + lane;
        if (e < E) {
            float s[P];
            float mx = -1e30f;
            #pragma unroll
            for (int p = 0; p < P; ++p) { s[p] = simsS[w][lane][p]; mx = fmaxf(mx, s[p]); }
            float sum = 0.0f;
            #pragma unroll
            for (int p = 0; p < P; ++p) { float ex = __expf(s[p] - mx); s[p] = ex; sum += ex; }
            const float inv = 1.0f / sum;
            float ent = 0.0f;
            float* pr = proba + (size_t)e * P;
            #pragma unroll
            for (int p = 0; p < P; ++p) {
                const float q = s[p] * inv;
                pr[p] = q;
                ent -= q * __logf(q + EPSE);
            }
            const float logit = -ent;
            logits[e] = logit;
            atomicMax(&segmaxE[dst[e]], fenc(logit));
        }
    }
}

// Pass 2: ex = exp(logit - segmax[dst]); atomic segment-sum of ex.
__global__ void edge_expsum_kernel(const float* __restrict__ logits,
                                   const int* __restrict__ dst,
                                   const unsigned* __restrict__ segmaxE,
                                   float* __restrict__ exb,
                                   float* __restrict__ segsum, int E) {
    int e = blockIdx.x * blockDim.x + threadIdx.x;
    if (e < E) {
        const int d = dst[e];
        const float ex = __expf(logits[e] - fdec(segmaxE[d]));
        exb[e] = ex;
        atomicAdd(&segsum[d], ex);
    }
}

// Pass 3: msg = proba @ multi_emb[edge_type]; scatter attn*msg into hidden[dst].
// One wave per edge: lane owns 4 of the 128 h-channels.
__global__ __launch_bounds__(256, 4)
void edge_message_kernel(const float* __restrict__ proba,
                         const float* __restrict__ exb,
                         const float* __restrict__ segsum,
                         const int* __restrict__ dst,
                         const int* __restrict__ etype,
                         const float* __restrict__ multi,   // [R][P][H], this layer
                         float* __restrict__ hout, int E) {
    const int w = threadIdx.x >> 5, lane = threadIdx.x & 31;
    const int e = blockIdx.x * 8 + w;
    if (e >= E) return;
    const int d  = dst[e];
    const int et = etype[e];
    const float attn = exb[e] / segsum[d];
    const float pv = (lane < P) ? proba[(size_t)e * P + lane] : 0.0f;
    const float* B = multi + (size_t)et * P * H + lane * 4;   // 1.2MB table, L2-hot
    float a0 = 0, a1 = 0, a2 = 0, a3 = 0;
    #pragma unroll
    for (int p = 0; p < P; ++p) {
        const float wp = __shfl(pv, p, 32);
        const float* bp = B + (size_t)p * H;
        a0 += wp * bp[0]; a1 += wp * bp[1]; a2 += wp * bp[2]; a3 += wp * bp[3];
    }
    float* o = hout + (size_t)d * H + lane * 4;
    atomicAdd(o + 0, attn * a0); atomicAdd(o + 1, attn * a1);
    atomicAdd(o + 2, attn * a2); atomicAdd(o + 3, attn * a3);
}

// Classifier: cosine sims of 1024 gathered rows vs last-layer prototypes.
__global__ void classifier_kernel(const float* __restrict__ hidden,
                                  const int* __restrict__ tails,
                                  const float* __restrict__ proto,  // [P][H] last layer
                                  float* __restrict__ out, int B) {
    const int w = threadIdx.x >> 5, lane = threadIdx.x & 31;
    const int row = blockIdx.x * 8 + w;
    if (row >= B) return;
    const float* hp = hidden + (size_t)tails[row] * H + lane * 4;
    const float h0 = hp[0], h1 = hp[1], h2 = hp[2], h3 = hp[3];
    float ss = h0 * h0 + h1 * h1 + h2 * h2 + h3 * h3;
    #pragma unroll
    for (int off = 16; off > 0; off >>= 1) ss += __shfl_xor(ss, off, 32);
    const float invh = 1.0f / fmaxf(sqrtf(ss), EPSN);
    for (int p = 0; p < P; ++p) {
        const float* pp = proto + (size_t)p * H + lane * 4;
        const float p0 = pp[0], p1 = pp[1], p2 = pp[2], p3 = pp[3];
        float dq = h0 * p0 + h1 * p1 + h2 * p2 + h3 * p3;
        float qq = p0 * p0 + p1 * p1 + p2 * p2 + p3 * p3;
        #pragma unroll
        for (int off = 16; off > 0; off >>= 1) {
            dq += __shfl_xor(dq, off, 32);
            qq += __shfl_xor(qq, off, 32);
        }
        if (lane == 0)
            out[(size_t)row * P + p] = dq * invh / fmaxf(sqrtf(qq), EPSN);
    }
}

extern "C" void kernel_launch(void* const* d_in, const int* in_sizes, int n_in,
                              void* d_out, int out_size, void* d_ws, size_t ws_size,
                              hipStream_t stream) {
    const int*   edge_index = (const int*)d_in[0];      // [2][E] int32
    const int*   etype      = (const int*)d_in[1];      // [E]
    const int*   tails      = (const int*)d_in[2];      // [1024]
    const float* multi      = (const float*)d_in[3];    // [L][R][P][H]
    const float* proto      = (const float*)d_in[4];    // [L][P][H]
    const float* srcEmb     = (const float*)d_in[5];    // [H]

    const int E     = in_sizes[0] / 2;
    const int L     = in_sizes[3] / (18 * P * H);
    const int Btail = in_sizes[2];
    const int* src = edge_index;
    const int* dst = edge_index + E;

    // Workspace: hidden ping-pong (2x25.6MB), proba [E][18] (57.6MB),
    // logits/ex [E], segmax/segsum [N]. ~116MB total, L2-resident.
    const size_t NH = (size_t)N_NODES * H;
    float*    hidA    = (float*)d_ws;
    float*    hidB    = hidA + NH;
    float*    proba   = hidB + NH;
    float*    logits  = proba + (size_t)E * P;
    float*    exb     = logits + E;
    unsigned* segmaxE = (unsigned*)(exb + E);
    float*    segsum  = (float*)(segmaxE + N_NODES);

    init_hidden_kernel<<<(int)((NH + 255) / 256), 256, 0, stream>>>(hidA, srcEmb, (int)NH);

    float* hin = hidA;
    float* hout = hidB;
    for (int l = 0; l < L; ++l) {
        layer_reset_kernel<<<(int)((NH + 255) / 256), 256, 0, stream>>>(
            hout, segmaxE, segsum, (int)NH, N_NODES);
        edge_sims_kernel<<<(E + 127) / 128, 256, 0, stream>>>(
            hin, proto + (size_t)l * P * H, src, dst, proba, logits, segmaxE, E);
        edge_expsum_kernel<<<(E + 255) / 256, 256, 0, stream>>>(
            logits, dst, segmaxE, exb, segsum, E);
        edge_message_kernel<<<(E + 7) / 8, 256, 0, stream>>>(
            proba, exb, segsum, dst, etype, multi + (size_t)l * 18 * P * H, hout, E);
        float* t = hin; hin = hout; hout = t;
    }

    classifier_kernel<<<(Btail + 7) / 8, 256, 0, stream>>>(
        hin, tails, proto + (size_t)(L - 1) * P * H, (float*)d_out, Btail);
}
